// PWCFlow_26560077758499
// MI455X (gfx1250) — matrix operand order repaired and checked by reading, not compiled, not run on hardware
//
#include <hip/hip_runtime.h>
#include <math.h>

// PWC-Net forward for MI455X (gfx1250, wave32).
// Heavy 3x3 convs -> implicit GEMM via v_wmma_f32_16x16x32_f16 (f16 in, f32 acc),
// weight tiles staged LDS-side by the Tensor Data Mover (tensor_load_to_lds,
// TENSORcnt-synchronized). Cost volume, warp, normalize, upsample, deconv and
// the 32->2 heads are bandwidth-trivial direct kernels.

typedef _Float16 half_t;
typedef _Float16 v16h __attribute__((ext_vector_type(16)));
typedef float    v8f  __attribute__((ext_vector_type(8)));
typedef unsigned int u32x4 __attribute__((ext_vector_type(4)));
typedef int          i32x4 __attribute__((ext_vector_type(4)));
typedef int          i32x8 __attribute__((ext_vector_type(8)));

#define LRELU_ALPHA 0.1f

#if defined(__has_builtin)
#if __has_builtin(__builtin_amdgcn_tensor_load_to_lds) && __has_builtin(__builtin_amdgcn_s_wait_tensorcnt)
#define USE_TDM 1
#endif
#endif
#ifndef USE_TDM
#define USE_TDM 0
#endif

static __device__ __forceinline__ int imin_(int a, int b) { return a < b ? a : b; }
static __device__ __forceinline__ int imax_(int a, int b) { return a > b ? a : b; }

// ---------------------------------------------------------------------------
// f32 -> f16 weight conversion
// ---------------------------------------------------------------------------
__global__ void f32_to_f16_kernel(const float* __restrict__ src,
                                  half_t* __restrict__ dst, int n) {
  int i = blockIdx.x * blockDim.x + threadIdx.x;
  if (i < n) dst[i] = (half_t)src[i];
}

// ---------------------------------------------------------------------------
// Implicit-GEMM 3x3 conv (SAME, optional dilation) with WMMA f16->f32.
//   M = B*H*W pixels, N = cout, K = cin*9.
//   X: NCHW with XC allocated channels (reads channels [0,cin)).
//   Y: NCHW with YC allocated channels, written at channel base ybase.
//   Wh: f16 weights, [cout][cin*9] (natural OIHW flattening).
// Workgroup: 128 threads = 4 waves; 64(M)x64(N) tile; K step 32.
// A tile: im2col gather (VALU). B tile: TDM tensor_load_to_lds with LDS
// padding producing the [64][40]-half layout (conflict-free, OOB zero-fill).
// ---------------------------------------------------------------------------
__global__ __launch_bounds__(128)
void conv_wmma_kernel(const float* __restrict__ X, int XC, int cin,
                      const half_t* __restrict__ Wh, const float* __restrict__ bias,
                      float* __restrict__ Y, int YC, int ybase, int cout,
                      int B, int H, int W, int dil, int do_leaky)
{
  const int HW   = H * W;
  const int M    = B * HW;
  const int Ktot = cin * 9;
  const int tile_m = blockIdx.x * 64;
  const int tile_n = blockIdx.y * 64;

  __shared__ half_t As[64][40];  // [pixel row 0..63][K 0..31] (+pad)
  __shared__ half_t Bs[64][40];  // [cout n 0..63][K 0..31] (+pad, matches TDM pad)

  const int tid   = threadIdx.x;
  const int wave  = tid >> 5;
  const int lane  = tid & 31;
  const int lhalf = lane >> 4;   // 0 / 1
  const int l15   = lane & 15;
  const int ka    = lhalf * 8;   // A-fragment K base per ISA layout

  // ---- hoisted pixel decode for the 16 A rows this thread stages ----
  // A staging index: idx = tid + i*128 -> row = wave + 4*i, kk = lane.
  int rowoff[16];  // b*XC*HW + y*W + x   (or <0 if row out of range)
  int rowy[16], rowx[16];
#pragma unroll
  for (int i = 0; i < 16; ++i) {
    int m = tile_m + wave + 4 * i;
    if (m < M) {
      int b = m / HW; int rem = m - b * HW;
      int y = rem / W; int x = rem - y * W;
      rowoff[i] = b * XC * HW + y * W + x;
      rowy[i] = y; rowx[i] = x;
    } else {
      rowoff[i] = -1; rowy[i] = 0; rowx[i] = 0;
    }
  }

  v8f acc[4];
#pragma unroll
  for (int j = 0; j < 4; ++j) acc[j] = (v8f){0,0,0,0,0,0,0,0};

  for (int k0 = 0; k0 < Ktot; k0 += 32) {
    // ---- stage A tile: one K-offset decode per thread per step ----
    {
      int kg = k0 + lane;
      int c  = kg / 9; int r = kg - c * 9;
      int ky = r / 3;  int kx = r - ky * 3;
      int dy = (ky - 1) * dil, dx = (kx - 1) * dil;
      int coff = c * HW + dy * W + dx;
      bool kok = (kg < Ktot);
#pragma unroll
      for (int i = 0; i < 16; ++i) {
        float v = 0.0f;
        int iy = rowy[i] + dy, ix = rowx[i] + dx;
        if (kok && rowoff[i] >= 0 && iy >= 0 && iy < H && ix >= 0 && ix < W)
          v = X[rowoff[i] + coff];
        As[wave + 4 * i][lane] = (half_t)v;
      }
    }

    // ---- stage B tile: Tensor Data Mover, 32(K) x 64(n) strided tile ----
#if USE_TDM
    if (tid < 32) {  // single wave issues the DMA; EXEC-independent
      unsigned long long gaddr =
          (unsigned long long)(const void*)(Wh + (size_t)tile_n * (size_t)Ktot + (size_t)k0);
      unsigned int ldsB = (unsigned int)(size_t)(void*)&Bs[0][0];
      int tdim0 = Ktot - k0;      // remaining K (OOB -> zero fill)
      int tdim1 = cout - tile_n;  // remaining cout rows (OOB -> zero fill)
      u32x4 g0;
      g0.x = 1u;                                  // count=1, user descriptor
      g0.y = ldsB;                                // lds_addr
      g0.z = (unsigned int)gaddr;                 // global_addr[31:0]
      g0.w = ((unsigned int)(gaddr >> 32) & 0x01FFFFFFu) | 0x80000000u; // [56:32] | type=2
      i32x8 g1;
      // data_size=1(2B), pad_enable, pad_interval=3(16 dwords), pad_amount=3(4 dwords)
      g1[0] = (1 << 16) | (1 << 20) | (3 << 22) | (3 << 25);
      g1[1] = (tdim0 & 0xFFFF) << 16;                               // tensor_dim0 lo
      g1[2] = ((tdim0 >> 16) & 0xFFFF) | ((tdim1 & 0xFFFF) << 16);  // dim0 hi | dim1 lo
      g1[3] = ((tdim1 >> 16) & 0xFFFF) | (32 << 16);                // dim1 hi | tile_dim0=32
      g1[4] = 64;                                                   // tile_dim1=64, tile_dim2=0
      g1[5] = Ktot;                                                 // tensor_dim0_stride lo
      g1[6] = 0;                                                    // stride hi | dim1_stride lo
      g1[7] = 0;
      i32x4 gz4 = (i32x4){0, 0, 0, 0};
#if __clang_major__ >= 23
      i32x8 gz8 = (i32x8){0, 0, 0, 0, 0, 0, 0, 0};
      __builtin_amdgcn_tensor_load_to_lds(g0, g1, gz4, gz4, gz8, 0);
#else
      __builtin_amdgcn_tensor_load_to_lds(g0, g1, gz4, gz4, 0);
#endif
      __builtin_amdgcn_s_wait_tensorcnt(0);
    }
#else
    // fallback: VALU staging into the same [n][k] layout
#pragma unroll
    for (int i = 0; i < 16; ++i) {
      int idx = tid + i * 128;
      int n = idx >> 5;
      int kk = idx & 31;
      int kg = k0 + kk;
      int nn = tile_n + n;
      half_t v = (half_t)0.0f;
      if (kg < Ktot && nn < cout) v = Wh[(size_t)nn * Ktot + kg];
      Bs[n][kk] = v;
    }
#endif
    __syncthreads();

    // ---- A fragment: lane holds row M=wave*16+l15; K contiguous (16*lhalf + e), matching B ----
    v16h a;
    const int mrow = wave * 16 + l15;
#pragma unroll
    for (int e = 0; e < 16; ++e) {
      int k = lhalf * 16 + e;
      a[e] = As[mrow][k];
    }
    // ---- 4 WMMAs: one 16x16 N-chunk each; B lane reads 16 contiguous halfs ----
#pragma unroll
    for (int j = 0; j < 4; ++j) {
      v16h bv;
#pragma unroll
      for (int e = 0; e < 16; ++e)
        bv[e] = Bs[j * 16 + l15][lhalf * 16 + e];
      acc[j] = __builtin_amdgcn_wmma_f32_16x16x32_f16(
          /*neg_a=*/false, a, /*neg_b=*/false, bv,
          /*c_mod=*/(short)0, acc[j], /*reuse_a=*/false, /*reuse_b=*/false);
    }
    __syncthreads();
  }

  // ---- epilogue: bias + optional LeakyReLU, scatter to NCHW ----
#pragma unroll
  for (int r = 0; r < 8; ++r) {
    int m = tile_m + wave * 16 + lhalf * 8 + r;
    if (m >= M) continue;
    int b = m / HW; int rem = m - b * HW;
    int y = rem / W; int x = rem - y * W;
    int pix = (b * YC) * HW + y * W + x;
#pragma unroll
    for (int j = 0; j < 4; ++j) {
      int n = tile_n + j * 16 + l15;
      if (n >= cout) continue;
      float v = acc[j][r] + bias[n];
      if (do_leaky) v = (v >= 0.0f) ? v : LRELU_ALPHA * v;
      Y[pix + (ybase + n) * HW] = v;
    }
  }
}

// ---------------------------------------------------------------------------
// Bilinear warp of f2 by flow (flow_to_warp + resample fused)
// ---------------------------------------------------------------------------
__global__ void warp_kernel(const float* __restrict__ data,
                            const float* __restrict__ flow,
                            float* __restrict__ out,
                            int B, int C, int H, int W)
{
  int idx = blockIdx.x * blockDim.x + threadIdx.x;
  int total = B * C * H * W;
  if (idx >= total) return;
  int x = idx % W; int t = idx / W;
  int y = t % H;   t /= H;
  int c = t % C;   int b = t / C;
  int HW = H * W;
  float wi = (float)y + flow[((b * 2 + 0) * H + y) * W + x];
  float wj = (float)x + flow[((b * 2 + 1) * H + y) * W + x];
  float i0 = floorf(wi), j0 = floorf(wj);
  float ai = wi - i0,    aj = wj - j0;
  const float* base = data + ((size_t)b * C + c) * HW;
  auto gather = [&](float ii, float jj) -> float {
    float valid = (ii >= 0.0f && ii <= (float)(H - 1) &&
                   jj >= 0.0f && jj <= (float)(W - 1)) ? 1.0f : 0.0f;
    int ic = (int)fminf(fmaxf(ii, 0.0f), (float)(H - 1));
    int jc = (int)fminf(fmaxf(jj, 0.0f), (float)(W - 1));
    return base[ic * W + jc] * valid;
  };
  float g00 = gather(i0,        j0);
  float g01 = gather(i0,        j0 + 1.0f);
  float g10 = gather(i0 + 1.0f, j0);
  float g11 = gather(i0 + 1.0f, j0 + 1.0f);
  out[idx] = g00 * (1.0f - ai) * (1.0f - aj) + g01 * (1.0f - ai) * aj +
             g10 * ai * (1.0f - aj)          + g11 * ai * aj;
}

// ---------------------------------------------------------------------------
// normalize_pair: per-(tensor,batch) sum/sumsq, combine, apply
// ---------------------------------------------------------------------------
__global__ void norm_stats_kernel(const float* __restrict__ f1,
                                  const float* __restrict__ f2,
                                  float* __restrict__ stats, int perBatch, int B)
{
  int g = blockIdx.x;            // 0..2B-1
  int t = g / B, b = g % B;
  const float* src = (t == 0 ? f1 : f2) + (size_t)b * perBatch;
  float s = 0.0f, ss = 0.0f;
  for (int i = threadIdx.x; i < perBatch; i += blockDim.x) {
    float v = src[i]; s += v; ss += v * v;
  }
  __shared__ float sh[256], sh2[256];
  sh[threadIdx.x] = s; sh2[threadIdx.x] = ss;
  __syncthreads();
  for (int off = blockDim.x / 2; off > 0; off >>= 1) {
    if ((int)threadIdx.x < off) {
      sh[threadIdx.x]  += sh[threadIdx.x + off];
      sh2[threadIdx.x] += sh2[threadIdx.x + off];
    }
    __syncthreads();
  }
  if (threadIdx.x == 0) { stats[g * 2] = sh[0]; stats[g * 2 + 1] = sh2[0]; }
}

__global__ void norm_finish_kernel(float* __restrict__ stats, int perBatch, int ngroups)
{
  if (blockIdx.x != 0 || threadIdx.x != 0) return;
  float N = (float)perBatch;
  float msum = 0.0f, vsum = 0.0f;
  for (int g = 0; g < ngroups; ++g) {
    float mu  = stats[g * 2] / N;
    float var = stats[g * 2 + 1] / N - mu * mu;
    msum += mu; vsum += var;
  }
  float m = msum / (float)ngroups;
  float v = vsum / (float)ngroups;
  stats[16] = m;
  stats[17] = 1.0f / sqrtf(v + 1e-16f);
}

__global__ void norm_apply_kernel(const float* __restrict__ src,
                                  float* __restrict__ dst,
                                  const float* __restrict__ stats, int n)
{
  int i = blockIdx.x * blockDim.x + threadIdx.x;
  if (i >= n) return;
  dst[i] = (src[i] - stats[16]) * stats[17];
}

// ---------------------------------------------------------------------------
// Cost volume (81 displacements, mean over 32 ch) + LeakyReLU fused
// ---------------------------------------------------------------------------
__global__ void costvol_kernel(const float* __restrict__ f1n,
                               const float* __restrict__ w2n,
                               float* __restrict__ cv, int B, int H, int W)
{
  int idx = blockIdx.x * blockDim.x + threadIdx.x;
  int total = B * 81 * H * W;
  if (idx >= total) return;
  int x = idx % W; int t = idx / W;
  int y = t % H;   t /= H;
  int d = t % 81;  int b = t / 81;
  int di = d / 9 - 4, dj = d % 9 - 4;
  int yy = y + di,   xx = x + dj;
  int HW = H * W;
  float s = 0.0f;
  if (yy >= 0 && yy < H && xx >= 0 && xx < W) {
    const float* p1 = f1n + (size_t)b * 32 * HW + y * W + x;
    const float* p2 = w2n + (size_t)b * 32 * HW + yy * W + xx;
#pragma unroll
    for (int c = 0; c < 32; ++c) s += p1[c * HW] * p2[c * HW];
  }
  s *= (1.0f / 32.0f);
  cv[idx] = (s >= 0.0f) ? s : LRELU_ALPHA * s;
}

// ---------------------------------------------------------------------------
// Channel-slice copy (src NCHW SC chans @sbase -> dst NCHW DC chans @dbase)
// ---------------------------------------------------------------------------
__global__ void copy_channels_kernel(const float* __restrict__ src, int SC, int sbase,
                                     float* __restrict__ dst, int DC, int dbase,
                                     int nc, int B, int HW)
{
  int idx = blockIdx.x * blockDim.x + threadIdx.x;
  int total = B * nc * HW;
  if (idx >= total) return;
  int p = idx % HW; int t = idx / HW;
  int c = t % nc;   int b = t / nc;
  dst[((size_t)b * DC + dbase + c) * HW + p] =
      src[((size_t)b * SC + sbase + c) * HW + p];
}

__global__ void copy_kernel(const float* __restrict__ s, float* __restrict__ d, int n)
{
  int i = blockIdx.x * blockDim.x + threadIdx.x;
  if (i < n) d[i] = s[i];
}

__global__ void add_kernel(const float* __restrict__ a, const float* __restrict__ b,
                           float* __restrict__ o, int n)
{
  int i = blockIdx.x * blockDim.x + threadIdx.x;
  if (i < n) o[i] = a[i] + b[i];
}

// ---------------------------------------------------------------------------
// Small direct 3x3 conv (used for 32->2 flow/refinement heads; no leaky)
// ---------------------------------------------------------------------------
__global__ void conv3x3_small_kernel(const float* __restrict__ X, int XC, int cin,
                                     const float* __restrict__ Wt,
                                     const float* __restrict__ bias,
                                     float* __restrict__ Y, int cout,
                                     int B, int H, int W)
{
  int idx = blockIdx.x * blockDim.x + threadIdx.x;
  int total = B * cout * H * W;
  if (idx >= total) return;
  int x = idx % W; int t = idx / W;
  int y = t % H;   t /= H;
  int o = t % cout; int b = t / cout;
  float s = bias[o];
  for (int c = 0; c < cin; ++c) {
#pragma unroll
    for (int ky = 0; ky < 3; ++ky) {
      int iy = y + ky - 1; if (iy < 0 || iy >= H) continue;
#pragma unroll
      for (int kx = 0; kx < 3; ++kx) {
        int ix = x + kx - 1; if (ix < 0 || ix >= W) continue;
        s += X[((b * XC + c) * H + iy) * W + ix] *
             Wt[((o * cin + c) * 3 + ky) * 3 + kx];
      }
    }
  }
  Y[((b * cout + o) * H + y) * W + x] = s;
}

// ---------------------------------------------------------------------------
// Transposed 4x4 stride-2 conv (context upsampler), 32->32
// w2[o][i][ky][kx] = w[i][o][3-ky][3-kx]
// ---------------------------------------------------------------------------
__global__ void deconv_kernel(const float* __restrict__ X,
                              const float* __restrict__ Wt,
                              const float* __restrict__ bias,
                              float* __restrict__ Y,
                              int B, int C, int H, int W)
{
  int H2 = 2 * H, W2 = 2 * W;
  int idx = blockIdx.x * blockDim.x + threadIdx.x;
  int total = B * C * H2 * W2;
  if (idx >= total) return;
  int ox = idx % W2; int t = idx / W2;
  int oy = t % H2;   t /= H2;
  int o = t % C;     int b = t / C;
  float s = bias[o];
  for (int ky = 0; ky < 4; ++ky) {
    int ty = oy + ky - 2;
    if (ty < 0 || (ty & 1)) continue;
    int iy = ty >> 1; if (iy >= H) continue;
    for (int kx = 0; kx < 4; ++kx) {
      int tx = ox + kx - 2;
      if (tx < 0 || (tx & 1)) continue;
      int ix = tx >> 1; if (ix >= W) continue;
      for (int c = 0; c < C; ++c)
        s += X[((b * C + c) * H + iy) * W + ix] *
             Wt[((c * C + o) * 4 + (3 - ky)) * 4 + (3 - kx)];
    }
  }
  Y[((b * C + o) * H2 + oy) * W2 + ox] = s;
}

// ---------------------------------------------------------------------------
// 2x bilinear upsample of flow (half-pixel centers) with *2.0 scale
// ---------------------------------------------------------------------------
__global__ void upsample_flow_kernel(const float* __restrict__ X,
                                     float* __restrict__ Y,
                                     int B, int H, int W)
{
  int H2 = 2 * H, W2 = 2 * W;
  int idx = blockIdx.x * blockDim.x + threadIdx.x;
  int total = B * 2 * H2 * W2;
  if (idx >= total) return;
  int ox = idx % W2; int t = idx / W2;
  int oy = t % H2;   t /= H2;
  int c = t % 2;     int b = t / 2;
  float sy = (oy + 0.5f) * 0.5f - 0.5f;
  float sx = (ox + 0.5f) * 0.5f - 0.5f;
  int y0 = (int)floorf(sy), x0 = (int)floorf(sx);
  float ay = sy - (float)y0, ax = sx - (float)x0;
  int y0c = imin_(imax_(y0, 0), H - 1), y1c = imin_(imax_(y0 + 1, 0), H - 1);
  int x0c = imin_(imax_(x0, 0), W - 1), x1c = imin_(imax_(x0 + 1, 0), W - 1);
  const float* p = X + ((size_t)b * 2 + c) * H * W;
  float v = p[y0c * W + x0c] * (1.0f - ay) * (1.0f - ax)
          + p[y0c * W + x1c] * (1.0f - ay) * ax
          + p[y1c * W + x0c] * ay * (1.0f - ax)
          + p[y1c * W + x1c] * ay * ax;
  Y[idx] = v * 2.0f;
}

// ---------------------------------------------------------------------------
// Host orchestration
// ---------------------------------------------------------------------------
extern "C" void kernel_launch(void* const* d_in, const int* in_sizes, int n_in,
                              void* d_out, int out_size, void* d_ws, size_t ws_size,
                              hipStream_t stream)
{
  (void)in_sizes; (void)n_in; (void)out_size; (void)ws_size;
  const int B = 4;
  static const int Hs[5] = {0, 128, 64, 32, 16};
  static const int Wd[5] = {0, 192, 96, 48, 24};
  static const int DEC[5] = {128, 128, 96, 64, 32};
  static const int RCH[6] = {128, 128, 128, 96, 64, 32};
  static const int RDIL[6] = {1, 2, 4, 8, 16, 1};

  auto P = [&](int i) { return (const float*)d_in[i]; };
  // params dict order: 8 feature tensors, then level1..level4 (14 each), refine (14)
  // level base = 8 + (L-1)*14 : w0,b0,...,w4,b4,wf,bf,wu,bu
  // refine base = 64          : w0,b0,...,w5,b5,wf,bf

  // bump allocator over workspace
  char* ws = (char*)d_ws;
  size_t off = 0;
  auto alloc = [&](size_t bytes) -> void* {
    void* p = ws + off;
    off += (bytes + 255) & ~(size_t)255;
    return p;
  };

  const int H1 = 128, W1 = 192, HW1 = H1 * W1;
  const int XC = 563;  // max dense-concat channels (147 + 128+128+96+64)

  float* x_cat   = (float*)alloc(sizeof(float) * (size_t)B * XC * HW1);
  float* warped2 = (float*)alloc(sizeof(float) * (size_t)B * 32 * HW1);
  float* f1n     = (float*)alloc(sizeof(float) * (size_t)B * 32 * HW1);
  float* w2n     = (float*)alloc(sizeof(float) * (size_t)B * 32 * HW1);
  float* cvb     = (float*)alloc(sizeof(float) * (size_t)B * 81 * HW1);
  float* context = (float*)alloc(sizeof(float) * (size_t)B * 32 * HW1);
  float* flow    = (float*)alloc(sizeof(float) * (size_t)B * 2 * HW1);
  float* flow_up = (float*)alloc(sizeof(float) * (size_t)B * 2 * HW1);
  float* ctx_up  = (float*)alloc(sizeof(float) * (size_t)B * 32 * HW1);
  float* rping   = (float*)alloc(sizeof(float) * (size_t)B * 128 * HW1);
  float* rpong   = (float*)alloc(sizeof(float) * (size_t)B * 128 * HW1);
  float* refout  = (float*)alloc(sizeof(float) * (size_t)B * 2 * HW1);
  float* stats   = (float*)alloc(sizeof(float) * 64);

  // ---- pre-convert all GEMM weights to f16 ----
  half_t* wh[4][5];
  for (int L = 1; L <= 4; ++L) {
    int cin = (L == 4) ? 113 : 147;
    for (int k = 0; k < 5; ++k) {
      int n = DEC[k] * cin * 9;
      wh[L - 1][k] = (half_t*)alloc(sizeof(half_t) * (size_t)n);
      f32_to_f16_kernel<<<(n + 255) / 256, 256, 0, stream>>>(
          P(8 + (L - 1) * 14 + 2 * k), wh[L - 1][k], n);
      cin += DEC[k];
    }
  }
  half_t* rwh[6];
  {
    int cin = 34;
    for (int k = 0; k < 6; ++k) {
      int n = RCH[k] * cin * 9;
      rwh[k] = (half_t*)alloc(sizeof(half_t) * (size_t)n);
      f32_to_f16_kernel<<<(n + 255) / 256, 256, 0, stream>>>(P(64 + 2 * k), rwh[k], n);
      cin = RCH[k];
    }
  }

  // output regions: flows (L1 refined, L2, L3, L4) concatenated
  const size_t oOff[5] = {0, 0, 196608, 245760, 258048};
  float* out = (float*)d_out;

  // ---- coarse-to-fine pyramid ----
  for (int L = 4; L >= 1; --L) {
    const int H = Hs[L], W = Wd[L], HW = H * W;
    const float* f1 = (const float*)d_in[2 * (L - 1)];
    const float* f2 = (const float*)d_in[2 * (L - 1) + 1];
    const int pb = 8 + (L - 1) * 14;

    const float* w2src;
    if (L == 4) {
      w2src = f2;
    } else {
      int tot = B * 32 * HW;
      warp_kernel<<<(tot + 255) / 256, 256, 0, stream>>>(f2, flow_up, warped2, B, 32, H, W);
      w2src = warped2;
    }

    const int perBatch = 32 * HW;
    norm_stats_kernel<<<2 * B, 256, 0, stream>>>(f1, w2src, stats, perBatch, B);
    norm_finish_kernel<<<1, 1, 0, stream>>>(stats, perBatch, 2 * B);
    {
      int n = B * perBatch;
      norm_apply_kernel<<<(n + 255) / 256, 256, 0, stream>>>(f1, f1n, stats, n);
      norm_apply_kernel<<<(n + 255) / 256, 256, 0, stream>>>(w2src, w2n, stats, n);
    }
    {
      int tot = B * 81 * HW;
      costvol_kernel<<<(tot + 255) / 256, 256, 0, stream>>>(f1n, w2n, cvb, B, H, W);
    }

    // build concat input
    int cin;
    if (L == 4) {
      cin = 113;
      int t1 = B * 81 * HW, t2 = B * 32 * HW;
      copy_channels_kernel<<<(t1 + 255) / 256, 256, 0, stream>>>(cvb, 81, 0, x_cat, XC, 0, 81, B, HW);
      copy_channels_kernel<<<(t2 + 255) / 256, 256, 0, stream>>>(f1, 32, 0, x_cat, XC, 81, 32, B, HW);
    } else {
      cin = 147;
      int t0 = B * 32 * HW, t1 = B * 2 * HW, t2 = B * 81 * HW;
      copy_channels_kernel<<<(t0 + 255) / 256, 256, 0, stream>>>(ctx_up, 32, 0, x_cat, XC, 0, 32, B, HW);
      copy_channels_kernel<<<(t1 + 255) / 256, 256, 0, stream>>>(flow_up, 2, 0, x_cat, XC, 32, 2, B, HW);
      copy_channels_kernel<<<(t2 + 255) / 256, 256, 0, stream>>>(cvb, 81, 0, x_cat, XC, 34, 81, B, HW);
      copy_channels_kernel<<<(t0 + 255) / 256, 256, 0, stream>>>(f1, 32, 0, x_cat, XC, 115, 32, B, HW);
    }

    // dense decoder: 5 WMMA convs, outputs appended into x_cat (last -> context)
    const int M = B * HW;
    for (int k = 0; k < 5; ++k) {
      int cout = DEC[k];
      float* Yp; int YC, ybase;
      if (k < 4) { Yp = x_cat; YC = XC; ybase = cin; }
      else       { Yp = context; YC = 32; ybase = 0; }
      dim3 grid((M + 63) / 64, (cout + 63) / 64);
      conv_wmma_kernel<<<grid, 128, 0, stream>>>(
          x_cat, XC, cin, wh[L - 1][k], P(pb + 2 * k + 1),
          Yp, YC, ybase, cout, B, H, W, 1, 1);
      cin += cout;
    }

    // flow head (32->2, no activation)
    {
      int tot = B * 2 * HW;
      conv3x3_small_kernel<<<(tot + 255) / 256, 256, 0, stream>>>(
          context, 32, 32, P(pb + 10), P(pb + 11), flow, 2, B, H, W);
      if (L < 4)
        add_kernel<<<(tot + 255) / 256, 256, 0, stream>>>(flow, flow_up, flow, tot);
      if (L > 1) {
        copy_kernel<<<(tot + 255) / 256, 256, 0, stream>>>(flow, out + oOff[L], tot);
        int tot2 = B * 2 * (2 * H) * (2 * W);
        upsample_flow_kernel<<<(tot2 + 255) / 256, 256, 0, stream>>>(flow, flow_up, B, H, W);
        int tot3 = B * 32 * (2 * H) * (2 * W);
        deconv_kernel<<<(tot3 + 255) / 256, 256, 0, stream>>>(
            context, P(pb + 12), P(pb + 13), ctx_up, B, 32, H, W);
      }
    }
  }

  // ---- refinement at level 1 ----
  {
    const int H = H1, W = W1, HW = HW1;
    int t0 = B * 32 * HW, t1 = B * 2 * HW;
    copy_channels_kernel<<<(t0 + 255) / 256, 256, 0, stream>>>(context, 32, 0, rping, 34, 0, 32, B, HW);
    copy_channels_kernel<<<(t1 + 255) / 256, 256, 0, stream>>>(flow, 2, 0, rping, 34, 32, 2, B, HW);
    int cin = 34;
    float* src = rping; float* dst = rpong;
    const int M = B * HW;
    for (int k = 0; k < 6; ++k) {
      int cout = RCH[k];
      dim3 grid((M + 63) / 64, (cout + 63) / 64);
      conv_wmma_kernel<<<grid, 128, 0, stream>>>(
          src, cin, cin, rwh[k], P(64 + 2 * k + 1),
          dst, cout, 0, cout, B, H, W, RDIL[k], 1);
      cin = cout;
      float* tmp = src; src = dst; dst = tmp;
    }
    // refinement head (32->2) then final flow = flow + refinement -> out[0]
    int tot = B * 2 * HW;
    conv3x3_small_kernel<<<(tot + 255) / 256, 256, 0, stream>>>(
        src, 32, 32, P(64 + 12), P(64 + 13), refout, 2, B, H, W);
    add_kernel<<<(tot + 255) / 256, 256, 0, stream>>>(flow, refout, out + oOff[1], tot);
  }
}
